// SINRNoSkip_21835613733176
// MI455X (gfx1250) — matrix operand order repaired and checked by reading, not compiled, not run on hardware
//
#include <hip/hip_runtime.h>

// ---------------- problem constants ----------------
#define B_      8
#define N_      32768
#define BN      (B_ * N_)          // 262144 points
#define HID     128
#define NFREQ   9
#define DEPTH   5
#define CODE    400
#define TILE_PTS 128               // points per workgroup (divides N_, 1 batch/block)
#define ZS      136                // f16 z row stride (272B rows, 16B aligned, bank-rotating)
#define WS      136                // f16 Wfwd row stride
#define FTS     24                 // f16 K-stride for Y / ftr operands (k=0..15 stored)

typedef __attribute__((ext_vector_type(16))) _Float16 v16h;
typedef __attribute__((ext_vector_type(8)))  _Float16 v8h;
typedef __attribute__((ext_vector_type(4)))  _Float16 v4h;
typedef __attribute__((ext_vector_type(8)))  float    v8f;

// ---------------------------------------------------------------------------
// Pre-kernel: code[i][b][h] = bfwd[i][h] + bcode[i][h] + sum_c Wcode[i][h][c]*a[b][c]
// ---------------------------------------------------------------------------
__global__ __launch_bounds__(128) void code_pre_kernel(
    const float* __restrict__ a, const float* __restrict__ Wcode,
    const float* __restrict__ bcode, const float* __restrict__ bfwd,
    float* __restrict__ code) {
  int i = blockIdx.x / B_;
  int b = blockIdx.x % B_;
  int h = threadIdx.x;
  const float4* w  = (const float4*)(Wcode + (size_t)(i * HID + h) * CODE);
  const float4* av = (const float4*)(a + b * CODE);
  float s = bcode[i * HID + h] + bfwd[i * HID + h];
#pragma unroll 4
  for (int c = 0; c < CODE / 4; ++c) {
    float4 wv = w[c], xv = av[c];
    s += wv.x * xv.x + wv.y * xv.y + wv.z * xv.z + wv.w * xv.w;
  }
  code[(i * B_ + b) * HID + h] = s;
}

// ---------------------------------------------------------------------------
// Main fused kernel. 512 threads = 16 waves. Wave w: col tile ct = w&7
// (cols 16*ct..16*ct+15), row group rp = w>>3 (rows 64*rp..64*rp+63 as four
// 16-row C tiles). All GEMMs (z@Wfwd, featurization Y@Wftr) go through
// v_wmma_f32_16x16x32_f16; modulation happens in C registers. Wfwd weights
// are register-prefetched one layer ahead to hide the L2 round trip.
// ---------------------------------------------------------------------------
__global__ __launch_bounds__(512) void sinr_kernel(
    const float* __restrict__ x, const float* __restrict__ Wftr,
    const float* __restrict__ bftr, const float* __restrict__ Wfwd,
    const float* __restrict__ Wout, const float* __restrict__ boutp,
    const float* __restrict__ code, float* __restrict__ out) {

  __shared__ _Float16 zsh[TILE_PTS * ZS];        // activations (f16 A operand)
  __shared__ _Float16 wsh[HID * WS];             // current layer Wfwd (f16 B operand)
  __shared__ _Float16 Yxsh[6 * TILE_PTS * FTS];  // [Y0..Y8,1,0..0] rows (A operand)
  __shared__ _Float16 ftrp[6 * HID * FTS];       // [Wftr row, bftr, 0..0] (B operand)
  __shared__ float    code_sh[DEPTH][HID];       // fused bfwd+bcode+Wcode@a
  __shared__ float    outsh[TILE_PTS][8];

  const int tid = threadIdx.x;
  const int p0  = blockIdx.x * TILE_PTS;
  const int b   = p0 >> 15;                      // batch index (N_ = 32768)

  // Prefetch layer-0 Wfwd into registers immediately (128B/thread).
  float4 wpre[8];
  {
    const float4* src = (const float4*)Wfwd;
#pragma unroll
    for (int j = 0; j < 8; ++j) wpre[j] = src[j * 512 + tid];
  }

  // ---------- phase A: stage constants, compute Y, echo x to output ----------
  // ftr operand: per (shift,col) 16 f16 slots: k0..8 = Wftr, k9 = bftr, k10..15 = 0
  for (int pr = tid; pr < 6 * HID; pr += 512) {
    int s = pr >> 7, c = pr & 127;
    const float* wrow = Wftr + (s * HID + c) * NFREQ;
    v8h lo, hi;
#pragma unroll
    for (int q = 0; q < 8; ++q) lo[q] = (_Float16)wrow[q];
    hi[0] = (_Float16)wrow[8];
    hi[1] = (_Float16)bftr[s * HID + c];
#pragma unroll
    for (int q = 2; q < 8; ++q) hi[q] = (_Float16)0.0f;
    _Float16* dst = &ftrp[(s * HID + c) * FTS];
    *(v8h*)dst = lo;
    *(v8h*)(dst + 8) = hi;
  }
  for (int k = tid; k < DEPTH * HID; k += 512) {
    int i = k >> 7, h = k & 127;
    code_sh[i][h] = code[(i * B_ + b) * HID + h];
  }
  if (tid < TILE_PTS / 2) {                      // x passthrough, 16B per thread
    ((float4*)(out + BN + p0 * 2))[tid] = ((const float4*)(x + p0 * 2))[tid];
  }
  // Y tasks: 128 points x 9 m-values = 1152 tasks.
  for (int T = tid; T < TILE_PTS * NFREQ; T += 512) {
    int pt   = T / NFREQ;
    int mIdx = T - pt * NFREQ;
    int m    = mIdx - 4;
    int am   = (m < 0) ? -m : m;
    float theta = x[(p0 + pt) * 2 + 0];
    float cphi  = cosf(x[(p0 + pt) * 2 + 1]);
    float somx2 = sqrtf(fmaxf(1.0f - cphi * cphi, 0.0f));
    float pmm = 1.0f;
    for (int i2 = 1; i2 <= am; ++i2) pmm *= -(2.0f * i2 - 1.0f) * somx2;
    float P[6];
    P[0] = pmm;
    float pb = cphi * (2.0f * am + 1.0f) * pmm;  // l = am+1
    P[1] = pb;
    float pa = pmm;
    for (int ll = am + 2; ll <= am + 5; ++ll) {
      float pll = ((2.0f * ll - 1.0f) * cphi * pb - (ll + am - 1.0f) * pa) /
                  (float)(ll - am);
      pa = pb; pb = pll;
      P[ll - am] = pll;
    }
    float trigmul = 1.0f;
    if (m != 0) {
      float tr   = (m > 0) ? cosf(am * theta) : sinf(am * theta);
      float sign = (am & 1) ? -1.0f : 1.0f;
      trigmul = sign * 1.41421356237309515f * tr;  // sqrt(2)
    }
    const float inv4pi = 0.07957747154594767f;
#pragma unroll
    for (int s = 0; s < 6; ++s) {
      int l = am + s;
      float ratio = 1.0f;                        // (l-am)! / (l+am)!
      for (int j = s + 1; j <= s + 2 * am; ++j) ratio /= (float)j;
      float nrm = sqrtf((2.0f * l + 1.0f) * inv4pi * ratio);
      _Float16* yr = &Yxsh[(s * TILE_PTS + pt) * FTS];
      yr[mIdx] = (_Float16)(nrm * P[s] * trigmul);
      if (mIdx == 0) {                           // one writer fills k=9..15
        yr[9] = (_Float16)1.0f;                  // bias slot
#pragma unroll
        for (int q = 10; q < 16; ++q) yr[q] = (_Float16)0.0f;
      }
    }
  }
  __syncthreads();

  const int lane   = tid & 31;
  const int laneLo = lane & 15;
  const int laneHi = lane >> 4;
  const int wave   = tid >> 5;
  const int ct     = wave & 7;                   // col tile
  const int rp     = wave >> 3;                  // row group (0/1): rows rp*64..+63
  const int col    = ct * 16 + laneLo;
  const int rbase  = rp * 64;

  // Featurization A fragment (K=32: lower 16 from LDS, upper 16 zero).
  auto ldYfrag = [&](int sh, int rowt) {
    v16h af = {};
    const _Float16* yr =
        &Yxsh[(sh * TILE_PTS + rbase + rowt * 16 + laneLo) * FTS + laneHi * 8];
    ((v8h*)&af)[0] = *(const v8h*)yr;            // k = laneHi*8 + 0..7
    return af;                                    // k = 16.. all zero
  };
  // Featurization B fragment (k = laneHi*8 + 0..7; k = 16.. all zero).
  auto ldFfrag = [&](int sh) {
    v16h bf = {};
    const _Float16* fr = &ftrp[(sh * HID + col) * FTS + laneHi * 8];
    ((v8h*)&bf)[0] = *(const v8h*)fr;
    return bf;
  };

  v8f zc;
#pragma unroll
  for (int r = 0; r < 8; ++r) zc[r] = 0.0f;

  // ---------- layer-0: z0 = Y(0)@Wftr0^T + b via WMMA, store f16 ----------
  {
    v16h bf = ldFfrag(0);
#pragma unroll
    for (int t = 0; t < 4; ++t) {
      v16h af = ldYfrag(0, t);
      v8f f = __builtin_amdgcn_wmma_f32_16x16x32_f16(false, af, false, bf,
                                                     (short)0, zc, false, false);
#pragma unroll
      for (int r = 0; r < 8; ++r) {
        int row = rbase + t * 16 + laneHi * 8 + r;
        zsh[row * ZS + col] = (_Float16)f[r];
      }
    }
  }

  // ---------- 5 layers: acc = code + z@Wfwd^T, modulate by Y(l+1)@Wftr^T ----------
  for (int layer = 0; layer < DEPTH; ++layer) {
    __syncthreads();   // z writes visible; previous wsh reads complete

    // commit prefetched Wfwd[layer] to LDS (f32 -> f16)
#pragma unroll
    for (int j = 0; j < 8; ++j) {
      int idx4 = j * 512 + tid;
      int e = idx4 * 4;
      int g = e >> 7, h = e & 127;
      v4h hv;
      hv[0] = (_Float16)wpre[j].x; hv[1] = (_Float16)wpre[j].y;
      hv[2] = (_Float16)wpre[j].z; hv[3] = (_Float16)wpre[j].w;
      *(v4h*)&wsh[g * WS + h] = hv;
    }
    // prefetch next layer's weights; loads stay in flight across this layer's WMMAs
    if (layer + 1 < DEPTH) {
      const float4* src = (const float4*)(Wfwd + (size_t)(layer + 1) * HID * HID);
#pragma unroll
      for (int j = 0; j < 8; ++j) wpre[j] = src[j * 512 + tid];
    }
    __syncthreads();

    // z-GEMM: 4 row tiles x 4 K chunks, accumulators seeded with fused code+bias
    float cv = code_sh[layer][col];
    v8f c[4];
#pragma unroll
    for (int t = 0; t < 4; ++t)
#pragma unroll
      for (int r = 0; r < 8; ++r) c[t][r] = cv;

#pragma unroll
    for (int kt = 0; kt < HID; kt += 32) {
      v16h bF;   // B: col, K = kt + laneHi*8 + [0..7] | kt+16+laneHi*8+[0..7]
      {
        const _Float16* wr = &wsh[col * WS + kt + laneHi * 8];
        ((v8h*)&bF)[0] = *(const v8h*)wr;
        ((v8h*)&bF)[1] = *(const v8h*)(wr + 16);
      }
#pragma unroll
      for (int t = 0; t < 4; ++t) {
        v16h aF;  // A: row, K = kt + laneHi*8 + [0..7] | kt+16+laneHi*8+[0..7]
        const _Float16* zr =
            &zsh[(rbase + t * 16 + laneLo) * ZS + kt + laneHi * 8];
        ((v8h*)&aF)[0] = *(const v8h*)zr;
        ((v8h*)&aF)[1] = *(const v8h*)(zr + 16);
        c[t] = __builtin_amdgcn_wmma_f32_16x16x32_f16(false, aF, false, bF,
                                                      (short)0, c[t], false, false);
      }
    }

    // F-GEMM (featurization for shift layer+1) straight into registers
    {
      int sh = layer + 1;
      v16h bf = ldFfrag(sh);
#pragma unroll
      for (int t = 0; t < 4; ++t) {
        v16h af = ldYfrag(sh, t);
        v8f f = __builtin_amdgcn_wmma_f32_16x16x32_f16(false, af, false, bf,
                                                       (short)0, zc, false, false);
#pragma unroll
        for (int r = 0; r < 8; ++r) c[t][r] *= f[r];   // modulate in registers
      }
    }
    __syncthreads();   // all zsh reads complete before writeback

    // writeback (f16 always; final layer's z is consumed from zsh too)
#pragma unroll
    for (int t = 0; t < 4; ++t)
#pragma unroll
      for (int r = 0; r < 8; ++r) {
        int row = rbase + t * 16 + laneHi * 8 + r;
        zsh[row * ZS + col] = (_Float16)c[t][r];
      }
  }
  __syncthreads();

  // ---------- output: out[p] = z_final[p] . Wout + bout ----------
  {
    int n = tid >> 3, seg = tid & 7;
    const _Float16* zr = &zsh[n * ZS + seg * 16];
    v8h z0v = *(const v8h*)zr;
    v8h z1v = *(const v8h*)(zr + 8);
    float s = 0.0f;
#pragma unroll
    for (int j = 0; j < 8; ++j) {
      s += (float)z0v[j] * Wout[seg * 16 + j];
      s += (float)z1v[j] * Wout[seg * 16 + 8 + j];
    }
    outsh[n][seg] = s;
  }
  __syncthreads();
  if (tid < TILE_PTS) {
    float s = boutp[0];
#pragma unroll
    for (int j = 0; j < 8; ++j) s += outsh[tid][j];
    out[p0 + tid] = s;
  }
}

// ---------------------------------------------------------------------------
extern "C" void kernel_launch(void* const* d_in, const int* in_sizes, int n_in,
                              void* d_out, int out_size, void* d_ws, size_t ws_size,
                              hipStream_t stream) {
  const float* x     = (const float*)d_in[0];
  const float* a     = (const float*)d_in[1];
  const float* Wftr  = (const float*)d_in[2];
  const float* bftr  = (const float*)d_in[3];
  const float* Wfwd  = (const float*)d_in[4];
  const float* bfwd  = (const float*)d_in[5];
  const float* Wcode = (const float*)d_in[6];
  const float* bcode = (const float*)d_in[7];
  const float* Wout  = (const float*)d_in[8];
  const float* bout  = (const float*)d_in[9];
  float* out  = (float*)d_out;
  float* code = (float*)d_ws;                  // DEPTH*B_*HID floats = 20 KB

  code_pre_kernel<<<DEPTH * B_, HID, 0, stream>>>(a, Wcode, bcode, bfwd, code);
  sinr_kernel<<<BN / TILE_PTS, 512, 0, stream>>>(x, Wftr, bftr, Wfwd, Wout, bout,
                                                 code, out);
}
